// GlocalRefEncoderRNNCell_66443144069397
// MI455X (gfx1250) — matrix operand order repaired and checked by reading, not compiled, not run on hardware
//
#include <hip/hip_runtime.h>
#include <hip/hip_bf16.h>
#include <stdint.h>

#define NB 32
#define TL 512
#define GS 128
#define PD 256
#define BN_EPS 1e-5f

typedef __attribute__((ext_vector_type(16))) __bf16 v16bf;
typedef __attribute__((ext_vector_type(8)))  float  v8f;
typedef __attribute__((ext_vector_type(8)))  unsigned short v8us;

union Frag { v16bf bf; v8us us[2]; };

__device__ __forceinline__ __bf16 f2bf(float x) {
  union { float f; unsigned u; } a; a.f = x;
  unsigned r = a.u + 0x7FFFu + ((a.u >> 16) & 1u);
  return __builtin_bit_cast(__bf16, (unsigned short)(r >> 16));
}
__device__ __forceinline__ float bf2f(__bf16 b) {
  unsigned u = ((unsigned)__builtin_bit_cast(unsigned short, b)) << 16;
  return __builtin_bit_cast(float, u);
}
__device__ __forceinline__ float sigmoidf_(float x) { return 1.0f / (1.0f + __expf(-x)); }

// ---- WMMA fragment loaders (bf16 16x16x32, wave32 layouts per CDNA5 ISA 7.12.2) ----
// A (16x32, M=lane&15): lane<16 holds K {k0..k0+7, k0+16..k0+23}; lane>=16 offset +8.
__device__ __forceinline__ v16bf load_a_frag(const __bf16* row, int k0, int hi) {
  Frag f;
  f.us[0] = *(const v8us*)(row + k0 + hi * 8);
  f.us[1] = *(const v8us*)(row + k0 + 16 + hi * 8);
  return f.bf;
}
__device__ __forceinline__ v16bf load_a_frag_pred(const __bf16* row, int k0, int hi, bool ok) {
  v8us z = {0, 0, 0, 0, 0, 0, 0, 0};
  Frag f;
  f.us[0] = ok ? *(const v8us*)(row + k0 + hi * 8) : z;
  f.us[1] = ok ? *(const v8us*)(row + k0 + 16 + hi * 8) : z;
  return f.bf;
}
// B (32x16, N=lane&15): lane<16 holds K {k0..k0+7, k0+16..k0+23}; lane>=16 offset +8. Column n is a
// contiguous row of the [N,K] weight matrix -> two 16-byte loads.
__device__ __forceinline__ v16bf load_b_frag(const __bf16* col, int k0, int hi) {
  Frag f;
  f.us[0] = *(const v8us*)(col + k0 + hi * 8);
  f.us[1] = *(const v8us*)(col + k0 + 16 + hi * 8);
  return f.bf;
}

// ---- gfx1250 async copy: global -> LDS (tracked by ASYNCcnt) ----
__device__ __forceinline__ void async_load_b128(unsigned lds_off, const void* gaddr) {
  asm volatile("global_load_async_to_lds_b128 %0, %1, off"
               :: "v"(lds_off), "v"(gaddr)
               : "memory");
}
__device__ __forceinline__ void wait_async0() {
  asm volatile("s_wait_asynccnt 0x0" ::: "memory");
}

// =====================  weight / input packing  =====================

__global__ __launch_bounds__(256) void f2bf_kernel(const float* __restrict__ src,
                                                   __bf16* __restrict__ dst, int n) {
  int i = blockIdx.x * 256 + threadIdx.x;
  if (i < n) dst[i] = f2bf(src[i]);
}

// conv_ws [O][Ci2][3][3] fp32  ->  wp [tap(kh*3+kw)][O][Cpad] bf16 (zero padded)
__global__ __launch_bounds__(256) void convw_pack_kernel(const float* __restrict__ wsrc,
                                                         __bf16* __restrict__ wp,
                                                         int Cout, int Ci2, int Cpin) {
  int idx = blockIdx.x * 256 + threadIdx.x;
  int total = 9 * Cout * Cpin;
  if (idx >= total) return;
  int cp = idx % Cpin; int p = idx / Cpin;
  int o = p % Cout; int tap = p / Cout;
  int kh = tap / 3, kw = tap % 3;
  float v = (cp < Ci2) ? wsrc[(((size_t)o * Ci2 + cp) * 3 + kh) * 3 + kw] : 0.0f;
  wp[idx] = f2bf(v);
}

// inputs [N,80,512] fp32 -> act0 [N][512][80][32] bf16 (c0 = data, c1 = xx, c2 = yy, rest 0)
__global__ __launch_bounds__(256) void prep_kernel(const float* __restrict__ inp,
                                                   __bf16* __restrict__ act0) {
  long long idx = (long long)blockIdx.x * 256 + threadIdx.x;
  long long total = (long long)NB * TL * 80 * 32;
  if (idx >= total) return;
  int c = (int)(idx & 31); long long p = idx >> 5;
  int w = (int)(p % 80); long long q = p / 80;
  int h = (int)(q % TL); int n = (int)(q / TL);
  float v = 0.0f;
  if (c == 0)      v = inp[((size_t)n * 80 + w) * TL + h];
  else if (c == 1) v = -1.0f + 2.0f * (float)w / 79.0f;
  else if (c == 2) v = -1.0f + 2.0f * (float)h / 511.0f;
  act0[idx] = f2bf(v);
}

// =====================  conv (implicit GEMM via WMMA)  =====================
// actIn [N][512][Win][CPIN] bf16, wp [9][Cout][CPIN] bf16. One wave computes one 16x16
// output tile of y[M=N*512*Wout][Cout]; also accumulates per-channel BN sums.
template <int CPIN>
__global__ __launch_bounds__(256) void conv_wmma_kernel(const __bf16* __restrict__ actIn,
                                                        const __bf16* __restrict__ wp,
                                                        const float* __restrict__ bias,
                                                        __bf16* __restrict__ y,
                                                        float* __restrict__ s_sum,
                                                        float* __restrict__ s_sq,
                                                        int Win, int Wout, int Cout) {
  const int tid = threadIdx.x;
  const int wave = tid >> 5, lane = tid & 31;
  const int l16 = lane & 15, hi = lane >> 4;
  const int ntiles = Cout >> 4;
  const int M = NB * TL * Wout;
  const long long tiles = (long long)(M >> 4) * ntiles;
  long long tile = (long long)blockIdx.x * 8 + wave;
  if (tile >= tiles) return;               // wave-uniform exit; EXEC stays full for WMMA
  const int mt = (int)(tile / ntiles);
  const int nt = (int)(tile - (long long)mt * ntiles);

  const int m = mt * 16 + l16;
  const int w = m % Wout;
  const int h = (m / Wout) % TL;
  const int n = m / (Wout * TL);

  v8f acc = {0.f, 0.f, 0.f, 0.f, 0.f, 0.f, 0.f, 0.f};
  const __bf16* bcol = wp + (size_t)(nt * 16 + l16) * CPIN;
#pragma unroll
  for (int kh = 0; kh < 3; ++kh) {
    const int hy = h + kh - 1;
#pragma unroll
    for (int kw = 0; kw < 3; ++kw) {
      const int wx = 2 * w + kw - 1;
      const bool ok = ((unsigned)hy < (unsigned)TL) && ((unsigned)wx < (unsigned)Win);
      const __bf16* ap = actIn + ((size_t)((size_t)n * TL + hy) * Win + wx) * CPIN;
      const __bf16* bp = bcol + (size_t)(kh * 3 + kw) * Cout * CPIN;
#pragma unroll
      for (int k0 = 0; k0 < CPIN; k0 += 32) {
        v16bf a = load_a_frag_pred(ap, k0, hi, ok);
        v16bf b = load_b_frag(bp, k0, hi);
        acc = __builtin_amdgcn_wmma_f32_16x16x32_bf16(false, a, false, b, (short)0, acc,
                                                      false, false);
      }
    }
  }
  const int col = nt * 16 + l16;
  const float bi = bias[col];
  float ssum = 0.f, ssq = 0.f;
#pragma unroll
  for (int r = 0; r < 8; ++r) {
    const int mr = mt * 16 + r + 8 * hi;
    float v = acc[r] + bi;
    y[(size_t)mr * Cout + col] = f2bf(v);
    ssum += v; ssq += v * v;
  }
  atomicAdd(&s_sum[col], ssum);
  atomicAdd(&s_sq[col], ssq);
}

__global__ void bnstats_kernel(const float* __restrict__ s_sum, const float* __restrict__ s_sq,
                               const float* __restrict__ gamma, const float* __restrict__ beta,
                               float* __restrict__ scale, float* __restrict__ shift,
                               int C, float invM) {
  int c = threadIdx.x;
  if (c >= C) return;
  float m = s_sum[c] * invM;
  float v = s_sq[c] * invM - m * m;
  float sc = gamma[c] * rsqrtf(v + BN_EPS);
  scale[c] = sc; shift[c] = beta[c] - m * sc;
}

// y [M][Cout] bf16 -> actNext [N][512][Wout][Cnext] bf16 (BN+ReLU, coord channels, zero pad)
__global__ __launch_bounds__(256) void bnrelu_pack_kernel(const __bf16* __restrict__ y,
                                                          const float* __restrict__ scale,
                                                          const float* __restrict__ shift,
                                                          __bf16* __restrict__ actNext,
                                                          int Wout, int Cout, int Cnext) {
  long long idx = (long long)blockIdx.x * 256 + threadIdx.x;
  long long total = (long long)NB * TL * Wout * Cnext;
  if (idx >= total) return;
  int c = (int)(idx % Cnext);
  long long p = idx / Cnext;
  int w = (int)(p % Wout);
  long long q = p / Wout;
  int h = (int)(q % TL);
  float v;
  if (c < Cout) {
    float x = bf2f(y[(size_t)p * Cout + c]);
    v = fmaxf(x * scale[c] + shift[c], 0.0f);
  } else if (c == Cout)     v = -1.0f + 2.0f * (float)w / (float)(Wout - 1);
  else if (c == Cout + 1)   v = -1.0f + 2.0f * (float)h / (float)(TL - 1);
  else                      v = 0.0f;
  actNext[idx] = f2bf(v);
}

// layer-6 head: y [N*512*2][128] -> gruin [T][N][256], feat = c*2 + w (matches NCHW reshape)
__global__ __launch_bounds__(256) void final_pack_kernel(const __bf16* __restrict__ y,
                                                         const float* __restrict__ scale,
                                                         const float* __restrict__ shift,
                                                         __bf16* __restrict__ gruin) {
  long long idx = (long long)blockIdx.x * 256 + threadIdx.x;
  long long total = (long long)NB * TL * 256;
  if (idx >= total) return;
  int f = (int)(idx & 255);
  long long p = idx >> 8;
  int t = (int)(p % TL);
  int n = (int)(p / TL);
  int c = f >> 1, w = f & 1;
  float x = bf2f(y[(((size_t)n * TL + t) * 2 + w) * 128 + c]);
  float v = fmaxf(x * scale[c] + shift[c], 0.0f);
  gruin[((size_t)t * NB + n) * 256 + f] = f2bf(v);
}

// =====================  persistent WMMA recurrence  =====================
// GEMM: C[32][N] = A[32][K] * B^T (+bias), C to LDS. 8 waves split (N/16)*2 tiles evenly.
template <int K, int N>
__device__ __forceinline__ void gemm32(const __bf16* A, int ldA, const __bf16* B,
                                       const float* bias, float* C, int ldC,
                                       int wave, int l16, int hi) {
  const int ntiles = N >> 4;
  const int tiles = ntiles << 1;  // M = 32
  for (int t = wave; t < tiles; t += 8) {
    const int mt = t / ntiles;
    const int nt = t - mt * ntiles;
    const __bf16* arow = A + (size_t)(mt * 16 + l16) * ldA;
    const __bf16* bcol = B + (size_t)(nt * 16 + l16) * K;
    v8f acc = {0.f, 0.f, 0.f, 0.f, 0.f, 0.f, 0.f, 0.f};
#pragma unroll
    for (int k0 = 0; k0 < K; k0 += 32) {
      v16bf a = load_a_frag(arow, k0, hi);
      v16bf b = load_b_frag(bcol, k0, hi);
      acc = __builtin_amdgcn_wmma_f32_16x16x32_bf16(false, a, false, b, (short)0, acc,
                                                    false, false);
    }
    const int col = nt * 16 + l16;
    const float bi = bias[col];
#pragma unroll
    for (int r = 0; r < 8; ++r)
      C[(size_t)(mt * 16 + r + 8 * hi) * ldC + col] = acc[r] + bi;
  }
}

__global__ __launch_bounds__(256) void recurrence_kernel(
    const __bf16* __restrict__ gruin,                                   // [T][N][256]
    const __bf16* __restrict__ wihT, const __bf16* __restrict__ whhT,   // [384][256],[384][128]
    const float* __restrict__ bihT, const float* __restrict__ bhhT,
    const __bf16* __restrict__ wihG, const __bf16* __restrict__ whhG,   // [384][128] x2
    const float* __restrict__ bihG, const float* __restrict__ bhhG,
    const __bf16* __restrict__ w1, const float* __restrict__ b1,
    const float* __restrict__ g1, const float* __restrict__ be1,
    const __bf16* __restrict__ w2, const float* __restrict__ b2,
    const float* __restrict__ g2, const float* __restrict__ be2,
    float* __restrict__ out_t, float* __restrict__ out_g) {
  __shared__ float gi[NB * 384];
  __shared__ float gh[NB * 384];
  __shared__ __bf16 hts[NB * GS];
  __shared__ __bf16 hgs[NB * GS];
  __shared__ __bf16 xbuf[2][NB * 256];  // async-staged GRU inputs (double buffer, 2x16KB)
  float* tp = gi;  // [NB][PD] head pre-activations reuse gi/gh storage (dead by then)
  float* gp = gh;

  const int tid = threadIdx.x;
  const int wave = tid >> 5, lane = tid & 31;
  const int l16 = lane & 15, hi = lane >> 4;

  for (int i = tid; i < NB * GS; i += 256) {
    hts[i] = f2bf(0.0f);
    hgs[i] = f2bf(0.0f);
  }
  // stage x_0 into xbuf[0] with async global->LDS copies (16B per lane per op)
  for (int i = tid; i < NB * 256 / 8; i += 256)
    async_load_b128((unsigned)(size_t)&xbuf[0][i * 8], gruin + (size_t)i * 8);
  wait_async0();
  __syncthreads();

  for (int t = 0; t < TL; ++t) {
    // kick off async staging of x_{t+1} while step t computes
    if (t + 1 < TL) {
      const __bf16* xn = gruin + (size_t)(t + 1) * NB * 256;
      const int slot = (t + 1) & 1;
      for (int i = tid; i < NB * 256 / 8; i += 256)
        async_load_b128((unsigned)(size_t)&xbuf[slot][i * 8], xn + (size_t)i * 8);
    }
    if (t + 2 < TL)
      __builtin_prefetch(gruin + (size_t)(t + 2) * NB * 256 + (size_t)tid * 32, 0, 1);

    // temporal GRU gates (A for GEMM1 comes from LDS x-buffer)
    gemm32<256, 384>(&xbuf[t & 1][0], 256, wihT, bihT, gi, 384, wave, l16, hi);
    gemm32<128, 384>(hts, GS, whhT, bhhT, gh, 384, wave, l16, hi);
    __syncthreads();
    for (int i = tid; i < NB * GS; i += 256) {
      int n = i >> 7, j = i & 127;
      float r = sigmoidf_(gi[n * 384 + j] + gh[n * 384 + j]);
      float z = sigmoidf_(gi[n * 384 + 128 + j] + gh[n * 384 + 128 + j]);
      float nn = tanhf(gi[n * 384 + 256 + j] + r * gh[n * 384 + 256 + j]);
      float hp = bf2f(hts[i]);
      hts[i] = f2bf((1.0f - z) * nn + z * hp);
    }
    __syncthreads();

    // global GRU gates (input = detached ht; forward identical)
    gemm32<128, 384>(hts, GS, wihG, bihG, gi, 384, wave, l16, hi);
    gemm32<128, 384>(hgs, GS, whhG, bhhG, gh, 384, wave, l16, hi);
    __syncthreads();
    for (int i = tid; i < NB * GS; i += 256) {
      int n = i >> 7, j = i & 127;
      float r = sigmoidf_(gi[n * 384 + j] + gh[n * 384 + j]);
      float z = sigmoidf_(gi[n * 384 + 128 + j] + gh[n * 384 + 128 + j]);
      float nn = tanhf(gi[n * 384 + 256 + j] + r * gh[n * 384 + 256 + j]);
      float hp = bf2f(hgs[i]);
      hgs[i] = f2bf((1.0f - z) * nn + z * hp);
    }
    __syncthreads();

    // heads
    gemm32<128, 256>(hts, GS, w1, b1, tp, 256, wave, l16, hi);
    gemm32<128, 256>(hgs, GS, w2, b2, gp, 256, wave, l16, hi);
    __syncthreads();

    // BN1d over batch + ReLU + store; thread == feature
    {
      const int j = tid;
      float m = 0.f;
      for (int n = 0; n < NB; ++n) m += tp[n * 256 + j];
      m *= (1.0f / NB);
      float v = 0.f;
      for (int n = 0; n < NB; ++n) { float d = tp[n * 256 + j] - m; v += d * d; }
      v *= (1.0f / NB);
      float sc = g1[j] * rsqrtf(v + BN_EPS);
      float sh = be1[j] - m * sc;
      for (int n = 0; n < NB; ++n)
        out_t[((size_t)n * TL + t) * PD + j] = fmaxf(tp[n * 256 + j] * sc + sh, 0.0f);

      m = 0.f;
      for (int n = 0; n < NB; ++n) m += gp[n * 256 + j];
      m *= (1.0f / NB);
      v = 0.f;
      for (int n = 0; n < NB; ++n) { float d = gp[n * 256 + j] - m; v += d * d; }
      v *= (1.0f / NB);
      sc = g2[j] * rsqrtf(v + BN_EPS);
      sh = be2[j] - m * sc;
      for (int n = 0; n < NB; ++n)
        out_g[((size_t)n * TL + t) * PD + j] = fmaxf(gp[n * 256 + j] * sc + sh, 0.0f);
    }
    // retire this wave's async copies for x_{t+1}, then make them workgroup-visible
    wait_async0();
    __syncthreads();
  }
}

// =====================  host driver  =====================

static inline int cdiv64(long long a, long long b) { return (int)((a + b - 1) / b); }

extern "C" void kernel_launch(void* const* d_in, const int* in_sizes, int n_in,
                              void* d_out, int out_size, void* d_ws, size_t ws_size,
                              hipStream_t stream) {
  (void)in_sizes; (void)n_in; (void)out_size;
  const float* inp = (const float*)d_in[0];
  const float *convw[6], *convb[6], *bng[6], *bnb[6];
  for (int i = 0; i < 6; ++i) {
    convw[i] = (const float*)d_in[1 + i];
    convb[i] = (const float*)d_in[7 + i];
    bng[i] = (const float*)d_in[13 + i];
    bnb[i] = (const float*)d_in[19 + i];
  }

  static const int CI2[6]  = {3, 34, 34, 66, 66, 130};
  static const int CPIN[6] = {32, 64, 64, 96, 96, 160};
  static const int COUT[6] = {32, 32, 64, 64, 128, 128};
  static const int WIN[6]  = {80, 40, 20, 10, 5, 3};
  static const int WOUT[6] = {40, 20, 10, 5, 3, 2};
  static const int CNEXT[5] = {64, 64, 96, 96, 160};

  char* ws = (char*)d_ws;
  size_t off = 0;
  auto alloc = [&](size_t bytes) {
    size_t o = off;
    off += (bytes + 255) & ~(size_t)255;
    return o;
  };
  const size_t R0 = alloc((size_t)NB * TL * 80 * 32 * 2);   // act ping (max: layer-1 input)
  const size_t R1 = alloc((size_t)NB * TL * 40 * 64 * 2);   // act pong (max: layer-2 input)
  const size_t R2 = alloc((size_t)NB * TL * 40 * 32 * 2);   // pre-BN y (max: layer-1 out)
  const size_t GRUIN = alloc((size_t)TL * NB * 256 * 2);
  size_t WP[6];
  for (int l = 0; l < 6; ++l) WP[l] = alloc((size_t)9 * COUT[l] * CPIN[l] * 2);
  static const int WN[6] = {384 * 256, 384 * 128, 384 * 128, 384 * 128, 256 * 128, 256 * 128};
  size_t WG[6];
  for (int i = 0; i < 6; ++i) WG[i] = alloc((size_t)WN[i] * 2);
  const size_t STATS = alloc(4 * 128 * sizeof(float));
  if (off > ws_size) return;  // deterministic no-op if workspace is too small

  // pack conv weights  [tap][O][Cpad] bf16
  for (int l = 0; l < 6; ++l) {
    int total = 9 * COUT[l] * CPIN[l];
    convw_pack_kernel<<<cdiv64(total, 256), 256, 0, stream>>>(
        convw[l], (__bf16*)(ws + WP[l]), COUT[l], CI2[l], CPIN[l]);
  }
  // pack GRU / linear weights to bf16 (row-major [N][K] is already B-friendly)
  const float* wsrc[6] = {(const float*)d_in[25], (const float*)d_in[26],
                          (const float*)d_in[29], (const float*)d_in[30],
                          (const float*)d_in[33], (const float*)d_in[37]};
  for (int i = 0; i < 6; ++i)
    f2bf_kernel<<<cdiv64(WN[i], 256), 256, 0, stream>>>(wsrc[i], (__bf16*)(ws + WG[i]), WN[i]);

  // input + coord channels
  prep_kernel<<<cdiv64((long long)NB * TL * 80 * 32, 256), 256, 0, stream>>>(
      inp, (__bf16*)(ws + R0));

  float* s_sum = (float*)(ws + STATS);
  float* s_sq = s_sum + 128;
  float* s_scale = s_sum + 256;
  float* s_shift = s_sum + 384;
  const size_t actOff[2] = {R0, R1};

  for (int l = 0; l < 6; ++l) {
    const __bf16* actIn = (const __bf16*)(ws + actOff[l & 1]);
    __bf16* yb = (__bf16*)(ws + R2);
    hipMemsetAsync(ws + STATS, 0, 2 * 128 * sizeof(float), stream);
    const int M = NB * TL * WOUT[l];
    const long long tiles = (long long)(M / 16) * (COUT[l] / 16);
    const int blocks = cdiv64(tiles, 8);
    switch (CPIN[l]) {
      case 32:
        conv_wmma_kernel<32><<<blocks, 256, 0, stream>>>(actIn, (const __bf16*)(ws + WP[l]),
            convb[l], yb, s_sum, s_sq, WIN[l], WOUT[l], COUT[l]);
        break;
      case 64:
        conv_wmma_kernel<64><<<blocks, 256, 0, stream>>>(actIn, (const __bf16*)(ws + WP[l]),
            convb[l], yb, s_sum, s_sq, WIN[l], WOUT[l], COUT[l]);
        break;
      case 96:
        conv_wmma_kernel<96><<<blocks, 256, 0, stream>>>(actIn, (const __bf16*)(ws + WP[l]),
            convb[l], yb, s_sum, s_sq, WIN[l], WOUT[l], COUT[l]);
        break;
      default:
        conv_wmma_kernel<160><<<blocks, 256, 0, stream>>>(actIn, (const __bf16*)(ws + WP[l]),
            convb[l], yb, s_sum, s_sq, WIN[l], WOUT[l], COUT[l]);
        break;
    }
    bnstats_kernel<<<1, 128, 0, stream>>>(s_sum, s_sq, bng[l], bnb[l], s_scale, s_shift,
                                          COUT[l], 1.0f / (float)M);
    if (l < 5) {
      __bf16* actNext = (__bf16*)(ws + actOff[(l + 1) & 1]);
      long long total = (long long)NB * TL * WOUT[l] * CNEXT[l];
      bnrelu_pack_kernel<<<cdiv64(total, 256), 256, 0, stream>>>(
          yb, s_scale, s_shift, actNext, WOUT[l], COUT[l], CNEXT[l]);
    } else {
      long long total = (long long)NB * TL * 256;
      final_pack_kernel<<<cdiv64(total, 256), 256, 0, stream>>>(
          yb, s_scale, s_shift, (__bf16*)(ws + GRUIN));
    }
  }

  float* out_t = (float*)d_out;
  float* out_g = out_t + (size_t)NB * TL * PD;
  recurrence_kernel<<<1, 256, 0, stream>>>(
      (const __bf16*)(ws + GRUIN),
      (const __bf16*)(ws + WG[0]), (const __bf16*)(ws + WG[1]),
      (const float*)d_in[27], (const float*)d_in[28],
      (const __bf16*)(ws + WG[2]), (const __bf16*)(ws + WG[3]),
      (const float*)d_in[31], (const float*)d_in[32],
      (const __bf16*)(ws + WG[4]), (const float*)d_in[34],
      (const float*)d_in[35], (const float*)d_in[36],
      (const __bf16*)(ws + WG[5]), (const float*)d_in[38],
      (const float*)d_in[39], (const float*)d_in[40],
      out_t, out_g);
}
